// GraphReasoningLayer_37864431681721
// MI455X (gfx1250) — matrix operand order repaired and checked by reading, not compiled, not run on hardware
//
#include <hip/hip_runtime.h>
#include <hip/hip_bf16.h>
#include <math.h>

// ---------------------------------------------------------------------------
// Types for CDNA5 WMMA (wave32): v_wmma_f32_16x16x32_bf16
// ---------------------------------------------------------------------------
typedef __bf16 bf16_t;
typedef __attribute__((ext_vector_type(16))) bf16_t v16bf;
typedef __attribute__((ext_vector_type(8)))  bf16_t v8bf;
typedef __attribute__((ext_vector_type(8)))  float  v8f;

union AB16 { v16bf v; v8bf h[2]; };

__device__ __forceinline__ bf16_t to_bf16(float f) {
  unsigned u = __builtin_bit_cast(unsigned, f);
  u += 0x7FFFu + ((u >> 16) & 1u);   // round-to-nearest-even
  unsigned short s = (unsigned short)(u >> 16);
  return __builtin_bit_cast(bf16_t, s);
}

#define GN  4096
#define GD  768
#define GTD 2304
#define GH  4
#define GDH 192

// ---------------------------------------------------------------------------
// fp32 -> bf16 conversion (optionally also keep an fp32 copy)
// ---------------------------------------------------------------------------
__global__ void cvt_kernel(const float* __restrict__ in, bf16_t* __restrict__ ob,
                           float* __restrict__ of, int n) {
  int i = blockIdx.x * 256 + threadIdx.x;
  if (i < n) {
    float v = in[i];
    ob[i] = to_bf16(v);
    if (of) of[i] = v;
  }
}

// ---------------------------------------------------------------------------
// C[M,Nout] = A[M,768](bf16) @ W[Nout,768]^T (bf16) + bias (+ residual)
// block = 128 threads (4 waves); block tile = 32 rows x 256 cols.
// Each wave owns 32x64 (8 C fragments). K loop is software-pipelined:
// ping-pong fragment buffers so global_load_b128s for step k+32 are in
// flight while the WMMAs for step k execute.
// ---------------------------------------------------------------------------
__global__ void __launch_bounds__(128)
gemm_bf16_kernel(const bf16_t* __restrict__ A, const bf16_t* __restrict__ W,
                 const float* __restrict__ bias, const float* __restrict__ res,
                 float* __restrict__ outF, bf16_t* __restrict__ outB,
                 int M, int Nout) {
  const int K = GD;                       // every GEMM here has K = 768
  const int lane = threadIdx.x & 31;
  const int wave = threadIdx.x >> 5;
  const int h16 = lane >> 4, l16 = lane & 15;
  const int m0 = blockIdx.x * 32;
  const int n0 = blockIdx.y * 256 + wave * 64;

  v8f acc[2][4] = {};

  const bf16_t* arow0 = A + (long)(m0 + l16) * K + h16 * 16;       // rows m0..m0+15
  const bf16_t* arow1 = arow0 + (long)16 * K;                      // rows m0+16..m0+31
  const bf16_t* wrow[4];
#pragma unroll
  for (int t = 0; t < 4; ++t)
    wrow[t] = W + (long)(n0 + t * 16 + l16) * K + h16 * 16;

  AB16 a0[2], a1[2], bb[2][4];

#define LOAD_FRAGS(kk, buf)                                              \
  do {                                                                   \
    a0[buf].h[0] = *(const v8bf*)(arow0 + (kk));                         \
    a0[buf].h[1] = *(const v8bf*)(arow0 + (kk) + 8);                     \
    a1[buf].h[0] = *(const v8bf*)(arow1 + (kk));                         \
    a1[buf].h[1] = *(const v8bf*)(arow1 + (kk) + 8);                     \
    _Pragma("unroll")                                                    \
    for (int t = 0; t < 4; ++t) {                                        \
      bb[buf][t].h[0] = ((const v8bf*)(wrow[t] + (kk)))[0];              \
      bb[buf][t].h[1] = ((const v8bf*)(wrow[t] + (kk)))[1];              \
    }                                                                    \
  } while (0)

#define DO_WMMAS(buf)                                                    \
  do {                                                                   \
    _Pragma("unroll")                                                    \
    for (int t = 0; t < 4; ++t) {                                        \
      acc[0][t] = __builtin_amdgcn_wmma_f32_16x16x32_bf16(               \
          false, a0[buf].v, false, bb[buf][t].v, (short)0, acc[0][t],    \
          false, false);                                                 \
      acc[1][t] = __builtin_amdgcn_wmma_f32_16x16x32_bf16(               \
          false, a1[buf].v, false, bb[buf][t].v, (short)0, acc[1][t],    \
          false, false);                                                 \
    }                                                                    \
  } while (0)

  LOAD_FRAGS(0, 0);
  for (int k = 0; k < K; k += 64) {
    // L2 prefetch a cacheline's worth ahead on the streaming operands
    __builtin_prefetch(arow0 + k + 128, 0, 0);
    __builtin_prefetch(wrow[0] + k + 128, 0, 0);
    LOAD_FRAGS(k + 32, 1);     // K % 64 == 0  =>  k+32 always valid
    DO_WMMAS(0);
    int k2 = (k + 64 < K) ? k + 64 : 0;   // clamp: harmless redundant load on last iter
    LOAD_FRAGS(k2, 0);
    DO_WMMAS(1);
  }
#undef LOAD_FRAGS
#undef DO_WMMAS

#pragma unroll
  for (int g = 0; g < 2; ++g) {
#pragma unroll
    for (int t = 0; t < 4; ++t) {
      int col = n0 + t * 16 + l16;
      float bv = bias ? bias[col] : 0.f;
#pragma unroll
      for (int v = 0; v < 8; ++v) {
        int row = m0 + g * 16 + v + h16 * 8;
        long idx = (long)row * Nout + col;
        float val = acc[g][t][v] + bv;
        if (res)  val += res[idx];
        if (outF) outF[idx] = val;
        if (outB) outB[idx] = to_bf16(val);
      }
    }
  }
}

// ---------------------------------------------------------------------------
// Transpose the V section of qkv ([N, col 1536+d]) into vt[d][n] (bf16),
// so attention V B-fragments become two contiguous 16B loads along keys.
// ---------------------------------------------------------------------------
__global__ void transpose_v_kernel(const bf16_t* __restrict__ qkv, bf16_t* __restrict__ vt) {
  int n = blockIdx.x * 256 + threadIdx.x;
  int d = blockIdx.y;
  vt[(long)d * GN + n] = qkv[(long)n * GTD + 1536 + d];
}

// ---------------------------------------------------------------------------
// Flash attention, one wave per (16-query tile, head).
//   S^T = K_tile(16x192) x Q^T(192x16)  -> each lane owns ONE query's scores
//   online softmax (shfl_xor 16 reduction), P maps in-lane onto A-fragment
//   O += P(16x32) x V(32x192) over 12 dh tiles; normalize at end.
// ---------------------------------------------------------------------------
__global__ void __launch_bounds__(32)
flash_attn_kernel(const bf16_t* __restrict__ qkv, const bf16_t* __restrict__ vt,
                  const int* __restrict__ adj, bf16_t* __restrict__ oB) {
  const int lane = threadIdx.x & 31;
  const int h16 = lane >> 4, l16 = lane & 15;
  const int q0 = blockIdx.x * 16;
  const int hd = blockIdx.y;
  const float scale = 0.07216878364870323f;   // 1/sqrt(192)
  const float NEG_INF = -__builtin_inff();

  // Q as B-fragments (query = lane&15, dh chunks of 32)
  AB16 qb[6];
  {
    const bf16_t* qrow = qkv + (long)(q0 + l16) * GTD + hd * GDH + h16 * 16;
#pragma unroll
    for (int s = 0; s < 6; ++s) {
      qb[s].h[0] = ((const v8bf*)(qrow + s * 32))[0];
      qb[s].h[1] = ((const v8bf*)(qrow + s * 32))[1];
    }
  }

  v8f o[12] = {};
  float m_i = NEG_INF, l_i = 0.f;
  const int* adjrow = adj + (long)(q0 + l16) * GN;

  for (int j = 0; j < 128; ++j) {
    const int k032 = j * 32;
    float p[16];
    float tmax = NEG_INF;

#pragma unroll
    for (int t = 0; t < 2; ++t) {
      const int kt0 = k032 + t * 16;
      v8f s = {};
      const bf16_t* krow = qkv + (long)(kt0 + l16) * GTD + GD + hd * GDH + h16 * 16;
#pragma unroll
      for (int st = 0; st < 6; ++st) {
        AB16 a;
        a.h[0] = *(const v8bf*)(krow + st * 32);
        a.h[1] = *(const v8bf*)(krow + st * 32 + 8);
        s = __builtin_amdgcn_wmma_f32_16x16x32_bf16(false, a.v, false, qb[st].v,
                                                    (short)0, s, false, false);
      }
      const int4* ap = (const int4*)(adjrow + kt0 + h16 * 8);
      int4 a0 = ap[0], a1 = ap[1];
      int am[8] = {a0.x, a0.y, a0.z, a0.w, a1.x, a1.y, a1.z, a1.w};
#pragma unroll
      for (int v = 0; v < 8; ++v) {
        float sv = (am[v] > 0) ? s[v] * scale : NEG_INF;
        p[t * 8 + v] = sv;
        tmax = fmaxf(tmax, sv);
      }
    }

    tmax = fmaxf(tmax, __shfl_xor(tmax, 16, 32));
    float mnew = fmaxf(m_i, tmax);
    float alpha = (m_i == NEG_INF) ? 0.f : __expf(m_i - mnew);
    bool dead = (mnew == NEG_INF);
    float lsum = 0.f;
    AB16 pa;
#pragma unroll
    for (int i = 0; i < 16; ++i) {
      float pe = dead ? 0.f : __expf(p[i] - mnew);
      lsum += pe;
      pa.v[i] = to_bf16(pe);
    }
    lsum += __shfl_xor(lsum, 16, 32);
    l_i = l_i * alpha + lsum;
    m_i = mnew;

    // broadcast per-query rescale factor to the O C-fragment row mapping
    float al[8];
#pragma unroll
    for (int v = 0; v < 8; ++v) al[v] = __shfl(alpha, v + h16 * 8, 32);
#pragma unroll
    for (int nt = 0; nt < 12; ++nt)
#pragma unroll
      for (int v = 0; v < 8; ++v) o[nt][v] *= al[v];

    // O += P @ V  (V B-fragments contiguous from transposed vt)
#pragma unroll
    for (int nt = 0; nt < 12; ++nt) {
      AB16 b;
      const bf16_t* vp = vt + (long)(hd * GDH + nt * 16 + l16) * GN + k032 + h16 * 8;
      b.h[0] = ((const v8bf*)vp)[0];
      b.h[1] = ((const v8bf*)vp)[2];
      o[nt] = __builtin_amdgcn_wmma_f32_16x16x32_bf16(false, pa.v, false, b.v,
                                                      (short)0, o[nt], false, false);
    }
  }

  float linv[8];
#pragma unroll
  for (int v = 0; v < 8; ++v) {
    float lv = __shfl(l_i, v + h16 * 8, 32);
    linv[v] = 1.f / lv;
  }
#pragma unroll
  for (int nt = 0; nt < 12; ++nt)
#pragma unroll
    for (int v = 0; v < 8; ++v) {
      int row = q0 + v + h16 * 8;
      int col = hd * GDH + nt * 16 + l16;
      oB[(long)row * GD + col] = to_bf16(o[nt][v] * linv[v]);
    }
}

// ---------------------------------------------------------------------------
// LayerNorm (per-row, D=768) + exact GELU; writes fp32 x and bf16 x.
// ---------------------------------------------------------------------------
__global__ void __launch_bounds__(256)
ln_gelu_kernel(const float* __restrict__ h, const float* __restrict__ g,
               const float* __restrict__ b, float* __restrict__ xF,
               bf16_t* __restrict__ xB) {
  __shared__ float red[256];
  const int row = blockIdx.x;
  const int tid = threadIdx.x;
  const float* hr = h + (long)row * GD;
  float v0 = hr[tid], v1 = hr[tid + 256], v2 = hr[tid + 512];

  red[tid] = v0 + v1 + v2;
  __syncthreads();
  for (int st = 128; st > 0; st >>= 1) { if (tid < st) red[tid] += red[tid + st]; __syncthreads(); }
  float mu = red[0] * (1.f / 768.f);
  __syncthreads();

  float d0 = v0 - mu, d1 = v1 - mu, d2 = v2 - mu;
  red[tid] = d0 * d0 + d1 * d1 + d2 * d2;
  __syncthreads();
  for (int st = 128; st > 0; st >>= 1) { if (tid < st) red[tid] += red[tid + st]; __syncthreads(); }
  float rstd = rsqrtf(red[0] * (1.f / 768.f) + 1e-5f);

  float ds[3] = {d0, d1, d2};
  int   cs[3] = {tid, tid + 256, tid + 512};
#pragma unroll
  for (int e = 0; e < 3; ++e) {
    int c = cs[e];
    float y = ds[e] * rstd * g[c] + b[c];
    float ge = 0.5f * y * (1.f + erff(y * 0.7071067811865476f));
    xF[(long)row * GD + c] = ge;
    xB[(long)row * GD + c] = to_bf16(ge);
  }
}

// ---------------------------------------------------------------------------
// Host-side orchestration
// ---------------------------------------------------------------------------
extern "C" void kernel_launch(void* const* d_in, const int* in_sizes, int n_in,
                              void* d_out, int out_size, void* d_ws, size_t ws_size,
                              hipStream_t stream) {
  const int N = GN, D = GD, TD = GTD, HOPS = 3;
  const float* ent   = (const float*)d_in[0];
  const int*   adj   = (const int*)d_in[3];
  const float* w_qkv = (const float*)d_in[4];
  const float* b_qkv = (const float*)d_in[5];
  const float* w_ao  = (const float*)d_in[6];
  const float* b_ao  = (const float*)d_in[7];
  const float* w_lin = (const float*)d_in[8];
  const float* b_lin = (const float*)d_in[9];
  const float* ln_g  = (const float*)d_in[10];
  const float* ln_b  = (const float*)d_in[11];
  const float* w_pr  = (const float*)d_in[12];
  const float* b_pr  = (const float*)d_in[13];

  char* ws = (char*)d_ws;
  size_t off = 0;
  auto alloc = [&](size_t bytes) -> char* {
    char* p = ws + off;
    off += (bytes + 255) & ~(size_t)255;
    return p;
  };
  bf16_t* xB   = (bf16_t*)alloc((size_t)N * D * 2);
  bf16_t* yB   = (bf16_t*)alloc((size_t)N * D * 2);
  bf16_t* oB   = (bf16_t*)alloc((size_t)N * D * 2);
  bf16_t* qkvB = (bf16_t*)alloc((size_t)N * TD * 2);   // also reused as hF (fp32) later
  bf16_t* vtB  = (bf16_t*)alloc((size_t)D * N * 2);
  float*  xF   = (float*)alloc((size_t)N * D * 4);
  bf16_t* wQ   = (bf16_t*)alloc((size_t)HOPS * TD * D * 2);
  bf16_t* wO   = (bf16_t*)alloc((size_t)HOPS * D * D * 2);
  bf16_t* wL   = (bf16_t*)alloc((size_t)HOPS * D * D * 2);
  bf16_t* wP   = (bf16_t*)alloc((size_t)D * D * 2);
  float*  hF   = (float*)qkvB;  // qkv dead once GEMM3 runs; N*D*4 <= N*TD*2

  auto cvt = [&](const float* src, bf16_t* dstB, float* dstF, size_t n) {
    cvt_kernel<<<dim3((unsigned)((n + 255) / 256)), 256, 0, stream>>>(src, dstB, dstF, (int)n);
  };
  cvt(ent,   xB, xF,      (size_t)N * D);
  cvt(w_qkv, wQ, nullptr, (size_t)HOPS * TD * D);
  cvt(w_ao,  wO, nullptr, (size_t)HOPS * D * D);
  cvt(w_lin, wL, nullptr, (size_t)HOPS * D * D);
  cvt(w_pr,  wP, nullptr, (size_t)D * D);

  for (int hop = 0; hop < HOPS; ++hop) {
    // qkv = x @ Wq^T + b  (bf16 out)
    gemm_bf16_kernel<<<dim3(N / 32, TD / 256), 128, 0, stream>>>(
        xB, wQ + (size_t)hop * TD * D, b_qkv + hop * TD,
        nullptr, nullptr, qkvB, N, TD);
    // V -> vt[d][n]
    transpose_v_kernel<<<dim3(N / 256, D), 256, 0, stream>>>(qkvB, vtB);
    // masked flash attention -> oB (bf16)
    flash_attn_kernel<<<dim3(N / 16, GH), 32, 0, stream>>>(qkvB, vtB, adj, oB);
    // y = x + o @ Wo^T + b   (bf16 out only; fp32 residual from xF)
    gemm_bf16_kernel<<<dim3(N / 32, D / 256), 128, 0, stream>>>(
        oB, wO + (size_t)hop * D * D, b_ao + hop * D,
        xF, nullptr, yB, N, D);
    // h = y @ Wl^T + b  (fp32 out)
    gemm_bf16_kernel<<<dim3(N / 32, D / 256), 128, 0, stream>>>(
        yB, wL + (size_t)hop * D * D, b_lin + hop * D,
        nullptr, hF, nullptr, N, D);
    // x = gelu(layernorm(h))  -> xF (fp32 residual) + xB (bf16 for next GEMM)
    ln_gelu_kernel<<<N, 256, 0, stream>>>(hF, ln_g + hop * D, ln_b + hop * D, xF, xB);
  }
  // out = x @ Wp^T + b  (fp32)
  gemm_bf16_kernel<<<dim3(N / 32, D / 256), 128, 0, stream>>>(
      xB, wP, b_pr, nullptr, (float*)d_out, nullptr, N, D);
}
